// MambaBlock_43078521979254
// MI455X (gfx1250) — matrix operand order repaired and checked by reading, not compiled, not run on hardware
//
#include <hip/hip_runtime.h>

typedef unsigned short u16;
typedef unsigned int   u32;
typedef __attribute__((ext_vector_type(16))) __bf16 v16bf;
typedef __attribute__((ext_vector_type(8)))  float  v8f;
typedef __attribute__((ext_vector_type(8)))  u32    v8u;

#define BATCH   4
#define SEQLEN  4096
#define DMODEL  512
#define DSTATE  16
#define DINNER  1024
#define BS      (BATCH*SEQLEN)   // 16384 tokens

// ---------- helpers ----------
__device__ __forceinline__ u16 f2bf(float f) {
  u32 u = __builtin_bit_cast(u32, f);
  u32 r = u + 0x7FFFu + ((u >> 16) & 1u);   // round-to-nearest-even
  return (u16)(r >> 16);
}
__device__ __forceinline__ float sigm(float x) { return 1.f / (1.f + __expf(-x)); }

// ---------- weight pre-pack: fp32 row-major [K x N] -> bf16 WMMA B-fragments ----------
// frag id = nTile*(K/32) + kChunk; lane l holds col nTile*16+(l&15),
// K indices kChunk*32 + 16*(l>>4) + i, i=0..15 (contiguous k order, same as the A gather).
// Tiles padded to a multiple of 4 (zero-filled) so the GEMM inner loop is branch-free.
__global__ __launch_bounds__(256) void pack_weight(const float* __restrict__ W,
                                                   u16* __restrict__ Bp,
                                                   int K, int N, int nTilesPad) {
  int gid  = blockIdx.x * 256 + threadIdx.x;
  int lane = gid & 31;
  int frag = gid >> 5;
  int kChunks = K >> 5;
  if (frag >= nTilesPad * kChunks) return;
  int nTile = frag / kChunks;
  int kc    = frag - nTile * kChunks;
  int col   = nTile * 16 + (lane & 15);
  int kbase = kc * 32 + ((lane >> 4) << 4);
  u16* dst  = Bp + ((size_t)frag * 32 + lane) * 16;
#pragma unroll
  for (int i = 0; i < 16; ++i) {
    float v = (col < N) ? W[(size_t)(kbase + i) * N + col] : 0.f;
    dst[i] = f2bf(v);
  }
}

// ---------- LayerNorm -> bf16 ----------
__global__ __launch_bounds__(256) void ln_kernel(const float* __restrict__ x,
                                                 const float* __restrict__ g,
                                                 const float* __restrict__ b,
                                                 u16* __restrict__ xn) {
  int row = blockIdx.x, tid = threadIdx.x;
  const float* xr = x + (size_t)row * DMODEL;
  float v0 = xr[tid], v1 = xr[tid + 256];
  __shared__ float r1[256], r2[256];
  r1[tid] = v0 + v1;
  r2[tid] = v0 * v0 + v1 * v1;
  __syncthreads();
  for (int off = 128; off > 0; off >>= 1) {
    if (tid < off) { r1[tid] += r1[tid + off]; r2[tid] += r2[tid + off]; }
    __syncthreads();
  }
  float mu = r1[0] * (1.f / DMODEL);
  float var = r2[0] * (1.f / DMODEL) - mu * mu;
  float rs = rsqrtf(var + 1e-5f);
  u16* o = xn + (size_t)row * DMODEL;
  o[tid]       = f2bf((v0 - mu) * rs * g[tid]       + b[tid]);
  o[tid + 256] = f2bf((v1 - mu) * rs * g[tid + 256] + b[tid + 256]);
}

// ---------- bf16 WMMA GEMM:  C[M x N] = A[M x K](bf16) * Bp(packed bf16)  (+residual) ----------
// block = 256 thr = 8 waves as 4(M) x 2(N); wave computes a 32x32 tile: 2 A-frags x 2 B-frags
// = 4 WMMA per 32-K step, branch-free (B padded), double-buffered LDS A staging.
__global__ __launch_bounds__(256) void gemm_bf16_wmma(const u16* __restrict__ A,
                                                      const u16* __restrict__ Bp,
                                                      float* __restrict__ C,
                                                      const float* __restrict__ residual,
                                                      int N, int K) {
  __shared__ u32 lds0[128 * 20];            // 128 rows x 32 bf16, padded 20 dwords/row
  __shared__ u32 lds1[128 * 20];
  const int lane = threadIdx.x & 31;
  const int wave = threadIdx.x >> 5;
  const int wm   = wave >> 1;               // 0..3 : M sub-tile
  const int wn   = wave & 1;                // 0..1 : N sub-tile
  const int nTilesReal = N >> 4;
  const int kChunks    = K >> 5;            // 16 or 32 (even)
  const int tileB0 = blockIdx.x * 4 + wn * 2;
  const int tileB1 = tileB0 + 1;
  const int mBase  = blockIdx.y * 128;

  // cooperative A-tile staging: thread t loads 16 bf16 of row t>>1, half (t&1)
  const int ldRow = threadIdx.x >> 1;
  const int ldCol = (threadIdx.x & 1) << 4;
  const u16* aSrcBase = A + (size_t)(mBase + ldRow) * K + ldCol;
  const int  ldsOff   = ldRow * 20 + (ldCol >> 1);

  // A-fragment gather pattern (contiguous k = 16*(lane>>4) + s, same as packed B)
  const int rowSel  = lane & 15;
  const int halfSel = lane >> 4;            // 0 / 1
  const int aReadOff = (wm * 32 + rowSel) * 20 + halfSel * 8;

  v8f acc00 = {}, acc01 = {}, acc10 = {}, acc11 = {};

  auto stageLoad = [&](int kc, uint4& q0, uint4& q1) {
    const uint4* g = (const uint4*)(aSrcBase + (size_t)kc * 32);
    q0 = g[0]; q1 = g[1];
  };
  auto stageStore = [&](u32* buf, uint4 q0, uint4 q1) {
    uint4* d = (uint4*)(buf + ldsOff);
    d[0] = q0; d[1] = q1;
  };
  auto compute = [&](const u32* buf, int kc) {
    const u32* r0 = buf + aReadOff;
    const u32* r1 = r0 + 16 * 20;
    v8u au0, au1;
#pragma unroll
    for (int p = 0; p < 8; ++p) {
      int o = p;                            // K pairs {0..15} (+16 for hi half)
      au0[p] = r0[o];
      au1[p] = r1[o];
    }
    v16bf a0 = __builtin_bit_cast(v16bf, au0);
    v16bf a1 = __builtin_bit_cast(v16bf, au1);
    v8u bu0 = *(const v8u*)(Bp + (size_t)(tileB0 * kChunks + kc) * 512 + lane * 16);
    v8u bu1 = *(const v8u*)(Bp + (size_t)(tileB1 * kChunks + kc) * 512 + lane * 16);
    v16bf b0 = __builtin_bit_cast(v16bf, bu0);
    v16bf b1 = __builtin_bit_cast(v16bf, bu1);
    acc00 = __builtin_amdgcn_wmma_f32_16x16x32_bf16(false, a0, false, b0, (short)0, acc00, false, false);
    acc01 = __builtin_amdgcn_wmma_f32_16x16x32_bf16(false, a0, false, b1, (short)0, acc01, false, false);
    acc10 = __builtin_amdgcn_wmma_f32_16x16x32_bf16(false, a1, false, b0, (short)0, acc10, false, false);
    acc11 = __builtin_amdgcn_wmma_f32_16x16x32_bf16(false, a1, false, b1, (short)0, acc11, false, false);
  };

  uint4 q0, q1;
  stageLoad(0, q0, q1);
  stageStore(lds0, q0, q1);
  __syncthreads();

  for (int kc = 0; kc < kChunks; kc += 2) {
    uint4 p0, p1;
    stageLoad(kc + 1, p0, p1);              // kc+1 < kChunks always (kChunks even)
    compute(lds0, kc);
    stageStore(lds1, p0, p1);
    __syncthreads();

    bool more = (kc + 2) < kChunks;         // uniform
    if (more) stageLoad(kc + 2, q0, q1);
    compute(lds1, kc + 1);
    if (more) stageStore(lds0, q0, q1);
    __syncthreads();
  }

  const int colLane = lane & 15;
  const int rowOff  = halfSel << 3;
  const v8f accs[2][2] = {{acc00, acc01}, {acc10, acc11}};
#pragma unroll
  for (int i = 0; i < 2; ++i) {
#pragma unroll
    for (int j = 0; j < 2; ++j) {
      int tile = tileB0 + j;
      if (tile >= nTilesReal) continue;     // wave-uniform (padding tiles only)
      int col = tile * 16 + colLane;
#pragma unroll
      for (int r = 0; r < 8; ++r) {
        int row = mBase + wm * 32 + i * 16 + rowOff + r;
        size_t idx = (size_t)row * N + col;
        float v = accs[i][j][r];
        if (residual) v += residual[idx];
        C[idx] = v;
      }
    }
  }
}

// ---------- depthwise causal conv (D_CONV=4) + SiLU ; emits f32 + bf16 copies ----------
__global__ __launch_bounds__(256) void conv_silu(const float* __restrict__ xz,
                                                 const float* __restrict__ cw,
                                                 const float* __restrict__ cb,
                                                 float* __restrict__ xc_act,
                                                 u16* __restrict__ xc_bf) {
  int idx = blockIdx.x * 256 + threadIdx.x;      // over BS*DINNER
  int d   = idx & (DINNER - 1);
  int row = idx >> 10;
  int t   = row & (SEQLEN - 1);
  float a = cb[d];
#pragma unroll
  for (int j = 0; j < 4; ++j) {
    int tt = t - 3 + j;
    if (tt >= 0) a += cw[d * 4 + j] * xz[(size_t)(row - 3 + j) * (2 * DINNER) + d];
  }
  float s = a * sigm(a);
  xc_act[idx] = s;
  xc_bf[idx]  = f2bf(s);
}

// ---------- slice delta_raw out of x_dbl -> bf16 ----------
__global__ __launch_bounds__(256) void cvt_delta(const float* __restrict__ x_dbl,
                                                 u16* __restrict__ dr_bf) {
  int idx = blockIdx.x * 256 + threadIdx.x;
  int d = idx & (DINNER - 1);
  int row = idx >> 10;
  dr_bf[idx] = f2bf(x_dbl[(size_t)row * 1056 + d]);
}

// ---------- selective scan, fused softplus; lanes = (b, d, s), shuffle-reduce over s ----------
__global__ __launch_bounds__(256) void scan_kernel(const float* __restrict__ xc_act,
                                                   const float* __restrict__ x_dbl,
                                                   const float* __restrict__ d_lin,
                                                   const float* __restrict__ b_dt,
                                                   const float* __restrict__ A_log,
                                                   const float* __restrict__ D_par,
                                                   float* __restrict__ y) {
  int s = threadIdx.x & 15;
  int d = blockIdx.x * 16 + (threadIdx.x >> 4);
  int b = blockIdx.y;
  float Aneg = -expf(A_log[d * DSTATE + s]);
  float Dp   = D_par[d];
  float bdt  = b_dt[d];
  const float* xrow  = xc_act + (size_t)b * SEQLEN * DINNER + d;
  const float* dlrow = d_lin  + (size_t)b * SEQLEN * DINNER + d;
  const float* bc    = x_dbl  + (size_t)b * SEQLEN * 1056 + DINNER;  // B at +s, C at +16+s
  float* yrow        = y      + (size_t)b * SEQLEN * DINNER + d;
  float h = 0.f;
  for (int t = 0; t < SEQLEN; ++t) {
    float xt = xrow[(size_t)t * DINNER];
    float dl = dlrow[(size_t)t * DINNER] + bdt;
    float dt = (dl > 20.f) ? dl : log1pf(expf(dl));      // softplus
    float Bt = bc[(size_t)t * 1056 + s];
    float Ct = bc[(size_t)t * 1056 + 16 + s];
    float dA = __expf(dt * Aneg);
    h = dA * h + (dt * xt) * Bt;
    float p = h * Ct;
    p += __shfl_xor(p, 1, 32);
    p += __shfl_xor(p, 2, 32);
    p += __shfl_xor(p, 4, 32);
    p += __shfl_xor(p, 8, 32);                            // sum within 16-lane group
    if (s == 0) yrow[(size_t)t * DINNER] = p + Dp * xt;
  }
}

// ---------- y * silu(z) -> bf16 ----------
__global__ __launch_bounds__(256) void gate_kernel(const float* __restrict__ xz,
                                                   const float* __restrict__ y,
                                                   u16* __restrict__ yg_bf) {
  int idx = blockIdx.x * 256 + threadIdx.x;
  int d = idx & (DINNER - 1);
  int row = idx >> 10;
  float z = xz[(size_t)row * (2 * DINNER) + DINNER + d];
  yg_bf[idx] = f2bf(y[idx] * z * sigm(z));
}

// ---------- host ----------
extern "C" void kernel_launch(void* const* d_in, const int* in_sizes, int n_in,
                              void* d_out, int out_size, void* d_ws, size_t ws_size,
                              hipStream_t stream) {
  (void)in_sizes; (void)n_in; (void)out_size; (void)ws_size;
  const float* x      = (const float*)d_in[0];
  const float* W_in   = (const float*)d_in[1];
  const float* conv_w = (const float*)d_in[2];
  const float* conv_b = (const float*)d_in[3];
  const float* W_x    = (const float*)d_in[4];
  const float* W_dt   = (const float*)d_in[5];
  const float* b_dt   = (const float*)d_in[6];
  const float* A_log  = (const float*)d_in[7];
  const float* D_par  = (const float*)d_in[8];
  const float* W_out  = (const float*)d_in[9];
  const float* ln_g   = (const float*)d_in[10];
  const float* ln_b   = (const float*)d_in[11];
  float* out = (float*)d_out;

  char* ws = (char*)d_ws;
  size_t off = 0;
  auto alloc = [&](size_t bytes) -> char* {
    char* p = ws + off;
    off += (bytes + 255) & ~(size_t)255;
    return p;
  };
  // padded tile counts per GEMM (multiple of 4)
  const int tIn  = 2048 / 16;             // 128
  const int tX   = (66 + 3) & ~3;         // 68 (real 66)
  const int tDt  = 1024 / 16;             // 64
  const int tOut = 512 / 16;              // 32

  u16*   xn_bf  = (u16*)  alloc((size_t)BS * DMODEL * 2);
  float* xz     = (float*)alloc((size_t)BS * 2 * DINNER * 4);
  float* xc_act = (float*)alloc((size_t)BS * DINNER * 4);
  u16*   xc_bf  = (u16*)  alloc((size_t)BS * DINNER * 2);
  float* x_dbl  = (float*)alloc((size_t)BS * 1056 * 4);
  u16*   dr_bf  = (u16*)  alloc((size_t)BS * DINNER * 2);
  float* d_lin  = (float*)alloc((size_t)BS * DINNER * 4);
  float* ybuf   = (float*)alloc((size_t)BS * DINNER * 4);
  u16*   yg_bf  = (u16*)  alloc((size_t)BS * DINNER * 2);
  u16*   WpIn   = (u16*)  alloc((size_t)tIn  * (512  >> 5) * 512 * 2);
  u16*   WpX    = (u16*)  alloc((size_t)tX   * (1024 >> 5) * 512 * 2);
  u16*   WpDt   = (u16*)  alloc((size_t)tDt  * (1024 >> 5) * 512 * 2);
  u16*   WpOut  = (u16*)  alloc((size_t)tOut * (1024 >> 5) * 512 * 2);

  auto pack = [&](const float* W, u16* Bp, int K, int N, int nTilesPad) {
    int total = nTilesPad * (K >> 5) * 32;
    pack_weight<<<(total + 255) / 256, 256, 0, stream>>>(W, Bp, K, N, nTilesPad);
  };
  pack(W_in,  WpIn,  512,  2048, tIn);
  pack(W_x,   WpX,   1024, 1056, tX);
  pack(W_dt,  WpDt,  1024, 1024, tDt);
  pack(W_out, WpOut, 1024, 512,  tOut);

  ln_kernel<<<BS, 256, 0, stream>>>(x, ln_g, ln_b, xn_bf);

  // xz = xn @ W_in   [16384 x 2048], K=512
  gemm_bf16_wmma<<<dim3(tIn / 4, BS / 128), 256, 0, stream>>>(xn_bf, WpIn, xz, nullptr, 2048, 512);

  conv_silu<<<(BS * DINNER) / 256, 256, 0, stream>>>(xz, conv_w, conv_b, xc_act, xc_bf);

  // x_dbl = xc @ W_x [16384 x 1056], K=1024  (68 padded tiles, branch-free)
  gemm_bf16_wmma<<<dim3(tX / 4, BS / 128), 256, 0, stream>>>(xc_bf, WpX, x_dbl, nullptr, 1056, 1024);

  cvt_delta<<<(BS * DINNER) / 256, 256, 0, stream>>>(x_dbl, dr_bf);

  // d_lin = delta_raw @ W_dt [16384 x 1024], K=1024
  gemm_bf16_wmma<<<dim3(tDt / 4, BS / 128), 256, 0, stream>>>(dr_bf, WpDt, d_lin, nullptr, 1024, 1024);

  scan_kernel<<<dim3(DINNER / 16, BATCH), 256, 0, stream>>>(xc_act, x_dbl, d_lin, b_dt, A_log, D_par, ybuf);

  gate_kernel<<<(BS * DINNER) / 256, 256, 0, stream>>>(xz, ybuf, yg_bf);

  // out = x + yg @ W_out [16384 x 512], K=1024, residual fused
  gemm_bf16_wmma<<<dim3(tOut / 4, BS / 128), 256, 0, stream>>>(yg_bf, WpOut, out, x, 512, 1024);
}
